// FlowEmbeddingLayer_52561809768854
// MI455X (gfx1250) — matrix operand order repaired and checked by reading, not compiled, not run on hardware
//
#include <hip/hip_runtime.h>
#include <hip/hip_bf16.h>

// MI455X / gfx1250, wave32. Edge-MLP + scatter-max.
// Matrix work on v_wmma_f32_16x16x32_bf16; next-tile feature gather pipelined
// through GLOBAL_LOAD_ASYNC_TO_LDS_B128 (ASYNCcnt) when available.

typedef __attribute__((ext_vector_type(16))) __bf16 v16bf;
typedef __attribute__((ext_vector_type(8)))  __bf16 v8bf;
typedef __attribute__((ext_vector_type(4)))  __bf16 v4bf;
typedef __attribute__((ext_vector_type(8)))  float  v8f;

#if __has_builtin(__builtin_amdgcn_global_load_async_to_lds_b128)
#define HAS_ASYNC 1
// Builtin signature (from clang diagnostic): arg0 = 'int __vector_size(16) __device__ *'
// (global / AS1), arg1 = LDS side (AS3), then two int immediates (offset, cpol).
typedef int v4i_b __attribute__((vector_size(16)));
typedef __attribute__((address_space(1))) v4i_b* as1_v4i;
typedef __attribute__((address_space(3))) v4i_b* as3_v4i;
#else
#define HAS_ASYNC 0
#endif

#define WAVES   8
#define TPB     (WAVES * 32)
#define HDIM    128
#define FDIM    64
#define KPAD1   160      // 131 -> 5 chunks of K=32 (row 131 carries b1, rest 0)
#define NC1     5
#define NC2     4
#define MSGLD   160
#define GRID_MAIN 640

// ---- fragment loaders -------------------------------------------------------
__device__ __forceinline__ v16bf load_a_frag(const __bf16* rowBase, int chunk, int kh) {
  const __bf16* p = rowBase + chunk * 32 + kh * 8;
  v8bf lo = *(const v8bf*)(p);        // K = c*32 + kh*8 + [0..7]
  v8bf hi = *(const v8bf*)(p + 16);   // K = c*32 + 16 + kh*8 + [0..7]
  v16bf a;
#pragma unroll
  for (int j = 0; j < 8; ++j) { a[j] = lo[j]; a[j + 8] = hi[j]; }
  return a;
}

__device__ __forceinline__ v16bf load_b_frag(const __bf16* wbase, int region, int lane) {
  const __bf16* p = wbase + region * 512 + lane * 16;
  v8bf lo = *(const v8bf*)(p);
  v8bf hi = *(const v8bf*)(p + 8);
  v16bf b;
#pragma unroll
  for (int j = 0; j < 8; ++j) { b[j] = lo[j]; b[j + 8] = hi[j]; }
  return b;
}

// fp32 weights [Kreal x 128] -> bf16 LDS, pre-swizzled into B-fragment layout.
// Row `biasRow` (if >=0) is filled from `bias` (folds Linear bias into the GEMM).
__device__ __forceinline__ void stage_weight(const float* __restrict__ W,
                                             const float* __restrict__ bias,
                                             int Kreal, int biasRow, int Kpad,
                                             __bf16* dst, int tid) {
  const int nchunk = Kpad >> 5;
  const int total  = Kpad * HDIM;
  for (int idx = tid; idx < total; idx += TPB) {
    int k = idx >> 7;
    int n = idx & 127;
    float v = (k < Kreal) ? W[k * HDIM + n] : ((k == biasRow) ? bias[n] : 0.0f);
    int nt  = n >> 4;
    int ch  = k >> 5;
    int kin = k & 31;
    int ln  = (n & 15) | (((kin >> 3) & 1) << 4);
    int j   = (kin & 7) | ((kin >> 4) << 3);
    dst[(nt * nchunk + ch) * 512 + ln * 16 + j] = (__bf16)v;
  }
}

#if HAS_ASYNC
// Async DMA: next tile's gathered features (fp32) -> per-wave LDS staging buffer.
__device__ __forceinline__ void issue_prefetch(const float* __restrict__ x2f,
                                               const float* __restrict__ x1f,
                                               const int* __restrict__ edst,
                                               const int* __restrict__ esrc,
                                               int eg, int E, float* stage,
                                               int m, int kh) {
#pragma unroll
  for (int ep = 0; ep < 8; ++ep) {
    int eL = (ep << 1) | kh;
    int e = eg + eL; e = (e < E) ? e : (E - 1);       // clamp: idempotent under max
    int d = edst[e], s = esrc[e];
    v4i_b* g2 = (v4i_b*)(x2f + (long)d * FDIM + m * 4);   // 16B per lane
    v4i_b* g1 = (v4i_b*)(x1f + (long)s * FDIM + m * 4);
    v4i_b* l2 = (v4i_b*)(stage + eL * HDIM + m * 4);
    v4i_b* l1 = (v4i_b*)(stage + eL * HDIM + FDIM + m * 4);
    __builtin_amdgcn_global_load_async_to_lds_b128((as1_v4i)g2, (as3_v4i)l2, 0, 0);
    __builtin_amdgcn_global_load_async_to_lds_b128((as1_v4i)g1, (as3_v4i)l1, 0, 0);
  }
}
#endif

// ---- init: out[:N2*128] = 0, out[N2*128:] = x2_pos --------------------------
__global__ void flowemb_init_kernel(float* __restrict__ out,
                                    const float* __restrict__ x2p, int n2) {
  long total0 = (long)n2 * HDIM;
  long total1 = (long)n2 * 3;
  long stride = (long)gridDim.x * blockDim.x;
  for (long i = blockIdx.x * (long)blockDim.x + threadIdx.x; i < total0; i += stride)
    out[i] = 0.0f;
  for (long i = blockIdx.x * (long)blockDim.x + threadIdx.x; i < total1; i += stride)
    out[total0 + i] = x2p[i];
}

// ---- main kernel ------------------------------------------------------------
__global__ __launch_bounds__(TPB) void flowemb_mlp_kernel(
    const float* __restrict__ x1f, const float* __restrict__ x1p,
    const float* __restrict__ x2f, const float* __restrict__ x2p,
    const float* __restrict__ W1, const float* __restrict__ b1,
    const float* __restrict__ W2, const float* __restrict__ b2,
    const float* __restrict__ W3, const float* __restrict__ b3,
    const int* __restrict__ esrc, const int* __restrict__ edst,
    float* __restrict__ out, int E, int numTiles) {
  __shared__ __align__(16) __bf16 w1s[8 * NC1 * 512];       // 40960 B
  __shared__ __align__(16) __bf16 w2s[8 * NC2 * 512];       // 32768 B
  __shared__ __align__(16) __bf16 w3s[8 * NC2 * 512];       // 32768 B
  __shared__ float b2s[HDIM], b3s[HDIM];
  __shared__ __align__(16) __bf16 msgS[WAVES * 16 * MSGLD]; // 40960 B
  __shared__ __align__(16) __bf16 hS[WAVES * 16 * HDIM];    // 32768 B
  __shared__ __align__(16) int dstS[WAVES * 16];
#if HAS_ASYNC
  __shared__ __align__(16) float stageS[WAVES * 16 * HDIM]; // 65536 B fp32 staging
#endif

  const int tid = threadIdx.x;
  stage_weight(W1, b1, 2 * FDIM + 3, 131, KPAD1, w1s, tid);  // bias folded at K=131
  stage_weight(W2, b2, HDIM, -1, HDIM, w2s, tid);
  stage_weight(W3, b3, HDIM, -1, HDIM, w3s, tid);
  for (int i = tid; i < HDIM; i += TPB) { b2s[i] = b2[i]; b3s[i] = b3[i]; }
  __syncthreads();

  const int lane = tid & 31;
  const int wave = tid >> 5;
  const int m  = lane & 15;
  const int kh = lane >> 4;
  __bf16* msg = msgS + wave * 16 * MSGLD;
  __bf16* hb  = hS   + wave * 16 * HDIM;
  int*    dI  = dstS + wave * 16;
#if HAS_ASYNC
  float*  stg = stageS + wave * 16 * HDIM;
#endif

  // Constant part of msg rows, written once: col 131 = 1.0 (bias), 132.. = 0.
  if (lane < 16) {
    msg[lane * MSGLD + 131] = (__bf16)1.0f;
    for (int k = 132; k < KPAD1; ++k) msg[lane * MSGLD + k] = (__bf16)0.0f;
  }

  const int gw = blockIdx.x * WAVES + wave;
  const int nw = GRID_MAIN * WAVES;
  if (gw >= numTiles) return;

#if HAS_ASYNC
  issue_prefetch(x2f, x1f, edst, esrc, gw * 16, E, stg, m, kh);
#endif

  for (int tile = gw; tile < numTiles; tile += nw) {
    const int eg = tile * 16;
    int pf = tile + nw; pf = (pf < numTiles) ? pf : tile;   // uniform per wave

#if HAS_ASYNC
    asm volatile("s_wait_asynccnt 0" ::: "memory");  // tile's features in stg
    // convert fp32 staging -> bf16 msg (pure LDS traffic)
#pragma unroll
    for (int ep = 0; ep < 8; ++ep) {
      int eL = (ep << 1) | kh;
      const float4* sp = (const float4*)(stg + eL * HDIM + m * 8);
      float4 fa = sp[0], fb = sp[1];
      v8bf pk = {(__bf16)fa.x, (__bf16)fa.y, (__bf16)fa.z, (__bf16)fa.w,
                 (__bf16)fb.x, (__bf16)fb.y, (__bf16)fb.z, (__bf16)fb.w};
      *(v8bf*)(msg + eL * MSGLD + m * 8) = pk;
    }
#else
    // synchronous gather fallback
#pragma unroll
    for (int ep = 0; ep < 8; ++ep) {
      int eL = (ep << 1) | kh;
      int e = eg + eL; e = (e < E) ? e : (E - 1);
      int d = edst[e], s = esrc[e];
      float4 f2 = ((const float4*)x2f)[d * 16 + m];
      float4 f1 = ((const float4*)x1f)[s * 16 + m];
      __bf16* row = msg + eL * MSGLD;
      v4bf p2 = {(__bf16)f2.x, (__bf16)f2.y, (__bf16)f2.z, (__bf16)f2.w};
      v4bf p1 = {(__bf16)f1.x, (__bf16)f1.y, (__bf16)f1.z, (__bf16)f1.w};
      *(v4bf*)(row + m * 4)        = p2;
      *(v4bf*)(row + FDIM + m * 4) = p1;
    }
#endif
    if (lane < 16) {
      int e = eg + lane; e = (e < E) ? e : (E - 1);
      int d = edst[e], s = esrc[e];
      dI[lane] = d;
      __bf16* row = msg + lane * MSGLD;
      row[128] = (__bf16)(x1p[s * 3 + 0] - x2p[d * 3 + 0]);
      row[129] = (__bf16)(x1p[s * 3 + 1] - x2p[d * 3 + 1]);
      row[130] = (__bf16)(x1p[s * 3 + 2] - x2p[d * 3 + 2]);
    }

#if HAS_ASYNC
    asm volatile("s_wait_dscnt 0" ::: "memory");  // stg fully drained
    issue_prefetch(x2f, x1f, edst, esrc, pf * 16, E, stg, m, kh);  // overlap w/ WMMA
#endif

    // ---- layer 1: [16 x 160] @ [160 x 128]  (bias folded into K=131 row)
    v16bf a1[NC1];
#pragma unroll
    for (int c = 0; c < NC1; ++c) a1[c] = load_a_frag(msg + m * MSGLD, c, kh);
#pragma unroll
    for (int nt = 0; nt < 8; ++nt) {
      v8f acc = {0.f, 0.f, 0.f, 0.f, 0.f, 0.f, 0.f, 0.f};
#pragma unroll
      for (int c = 0; c < NC1; ++c) {
        v16bf b = load_b_frag(w1s, nt * NC1 + c, lane);
        acc = __builtin_amdgcn_wmma_f32_16x16x32_bf16(false, a1[c], false, b,
                                                      (short)0, acc, false, false);
      }
#pragma unroll
      for (int r = 0; r < 8; ++r) {
        float v = fmaxf(acc[r], 0.0f);
        hb[(r + 8 * kh) * HDIM + nt * 16 + m] = (__bf16)v;
      }
    }

    // ---- layer 2: [16 x 128] @ [128 x 128]; h2 into msg rows (cols 0..127)
    v16bf a2[NC2];
#pragma unroll
    for (int c = 0; c < NC2; ++c) a2[c] = load_a_frag(hb + m * HDIM, c, kh);
#pragma unroll
    for (int nt = 0; nt < 8; ++nt) {
      v8f acc = {0.f, 0.f, 0.f, 0.f, 0.f, 0.f, 0.f, 0.f};
#pragma unroll
      for (int c = 0; c < NC2; ++c) {
        v16bf b = load_b_frag(w2s, nt * NC2 + c, lane);
        acc = __builtin_amdgcn_wmma_f32_16x16x32_bf16(false, a2[c], false, b,
                                                      (short)0, acc, false, false);
      }
      float bias = b2s[nt * 16 + m];
#pragma unroll
      for (int r = 0; r < 8; ++r) {
        float v = fmaxf(acc[r] + bias, 0.0f);
        msg[(r + 8 * kh) * MSGLD + nt * 16 + m] = (__bf16)v;
      }
    }

    // ---- layer 3 + scatter-max (vals >= 0, zero-inited out -> int atomic max)
    v16bf a3[NC2];
#pragma unroll
    for (int c = 0; c < NC2; ++c) a3[c] = load_a_frag(msg + m * MSGLD, c, kh);

    // hoist this lane's 8 dst indices (two ds_load_b128) + output bases
    long ob[8];
    {
      const int4* q = (const int4*)(dI + 8 * kh);
      int4 qa = q[0], qb = q[1];
      int dd[8] = {qa.x, qa.y, qa.z, qa.w, qb.x, qb.y, qb.z, qb.w};
#pragma unroll
      for (int r = 0; r < 8; ++r) ob[r] = (long)dd[r] * HDIM + m;
    }
#pragma unroll
    for (int nt = 0; nt < 8; ++nt) {
      v8f acc = {0.f, 0.f, 0.f, 0.f, 0.f, 0.f, 0.f, 0.f};
#pragma unroll
      for (int c = 0; c < NC2; ++c) {
        v16bf b = load_b_frag(w3s, nt * NC2 + c, lane);
        acc = __builtin_amdgcn_wmma_f32_16x16x32_bf16(false, a3[c], false, b,
                                                      (short)0, acc, false, false);
      }
      float bias = b3s[nt * 16 + m];
#pragma unroll
      for (int r = 0; r < 8; ++r) {
        float v = fmaxf(acc[r] + bias, 0.0f);
        atomicMax((int*)(out + ob[r] + nt * 16), __float_as_int(v));
      }
    }
  }
}

// ---- host launch ------------------------------------------------------------
extern "C" void kernel_launch(void* const* d_in, const int* in_sizes, int n_in,
                              void* d_out, int out_size, void* d_ws, size_t ws_size,
                              hipStream_t stream) {
  const float* x1f = (const float*)d_in[0];
  const float* x1p = (const float*)d_in[1];
  const float* x2f = (const float*)d_in[2];
  const float* x2p = (const float*)d_in[3];
  const float* W1  = (const float*)d_in[4];
  const float* b1  = (const float*)d_in[5];
  const float* W2  = (const float*)d_in[6];
  const float* b2  = (const float*)d_in[7];
  const float* W3  = (const float*)d_in[8];
  const float* b3  = (const float*)d_in[9];
  const int* esrc  = (const int*)d_in[10];
  const int* edst  = (const int*)d_in[11];
  float* out = (float*)d_out;

  int E  = in_sizes[10];
  int N2 = in_sizes[3] / 3;
  int numTiles = (E + 15) / 16;

  flowemb_init_kernel<<<dim3(2048), dim3(256), 0, stream>>>(out, x2p, N2);
  flowemb_mlp_kernel<<<dim3(GRID_MAIN), dim3(TPB), 0, stream>>>(
      x1f, x1p, x2f, x2p, W1, b1, W2, b2, W3, b3, esrc, edst, out, E, numTiles);
}
